// AGTLayer_7980049236043
// MI455X (gfx1250) — matrix operand order repaired and checked by reading, not compiled, not run on hardware
//
#include <hip/hip_runtime.h>
#include <stdint.h>

// Problem constants (match reference)
#define B_  4
#define N_  2048
#define D_  1024
#define H_  8
#define HD_ 128
#define M_  (B_ * N_)   // 8192 rows

// GEMM tiling
#define KSTEP    64
#define LDS_ROW  72                      // bf16 elems per LDS row: 64 + 8 pad (144 B)
#define A_BYTES  (128 * LDS_ROW * 2)     // 18432 B per tile (A or B)
#define BUF_BYTES (2 * A_BYTES)          // A tile + B tile per buffer
#define SMEM_BYTES (2 * BUF_BYTES)       // double buffered: 73728 B

typedef __attribute__((ext_vector_type(16))) __bf16 v16bf;
typedef __attribute__((ext_vector_type(8)))  __bf16 v8bf;
typedef __attribute__((ext_vector_type(4)))  __bf16 v4bf;
typedef __attribute__((ext_vector_type(8)))  float  v8f;
typedef __attribute__((ext_vector_type(4)))  unsigned int v4u;
typedef __attribute__((ext_vector_type(8)))  int v8i;
typedef __attribute__((ext_vector_type(4)))  int v4i;

static __device__ __forceinline__ float leaky(float x) {
    return x > 0.0f ? x : 0.01f * x;
}

// ---------------------------------------------------------------------------
// K1: f32 -> bf16 conversion (one float4 per thread)
// ---------------------------------------------------------------------------
__global__ __launch_bounds__(256) void cvt_bf16_kernel(const float* __restrict__ src,
                                                       __bf16* __restrict__ dst, int n4) {
    int i = blockIdx.x * blockDim.x + threadIdx.x;
    if (i < n4) {
        float4 f = ((const float4*)src)[i];
        v4bf o;
        o.x = (__bf16)f.x; o.y = (__bf16)f.y; o.z = (__bf16)f.z; o.w = (__bf16)f.w;
        ((v4bf*)dst)[i] = o;
    }
}

// ---------------------------------------------------------------------------
// TDM: issue one 128 x KSTEP bf16 tile load  (global -> LDS, padded rows)
// D# group0/group1 per CDNA5 ISA.
// Toolchain uses the 6-arg builtin: (u32x4, i32x8, i32x4, i32x4, i32x8, i32).
// ---------------------------------------------------------------------------
static __device__ __forceinline__ void tdm_issue_tile(unsigned lds_byte,
                                                      const void* gaddr,
                                                      unsigned nrows) {
    unsigned long long ga = (unsigned long long)(uintptr_t)gaddr;
    v4u g0;
    g0[0] = 1u;                                         // count=1, no gather
    g0[1] = lds_byte;                                   // LDS dest byte address
    g0[2] = (unsigned)ga;                               // global_addr[31:0]
    g0[3] = (unsigned)((ga >> 32) & 0x01FFFFFFull)      // global_addr[56:32]
          | (2u << 30);                                 // type = 2 (image)
    v8i g1;
    g1[0] = (int)((1u << 16)      // data_size: 2 bytes
                | (1u << 20)      // pad_enable
                | (4u << 22)      // pad_interval: 32 DWORDs = 128 B (one row)
                | (3u << 25));    // pad_amount: 4 DWORDs = 16 B
    g1[1] = (int)(((unsigned)D_ & 0xFFFFu) << 16);                   // tensor_dim0 lo16
    g1[2] = (int)(((nrows & 0xFFFFu) << 16) | ((unsigned)D_ >> 16)); // dim1 lo16 | dim0 hi16
    g1[3] = (int)(((unsigned)KSTEP << 16) | (nrows >> 16));          // tile_dim0 | dim1 hi16
    g1[4] = 128;                                                     // tile_dim1 (rows)
    g1[5] = D_;                                                      // tensor_dim0_stride lo32
    g1[6] = 0;
    g1[7] = 0;
    v4i z4 = {0, 0, 0, 0};                   // groups 2/3: unused (2D tensor)
    v8i z8 = {0, 0, 0, 0, 0, 0, 0, 0};
    __builtin_amdgcn_tensor_load_to_lds(g0, g1, z4, z4, z8, 0);
}

// ---------------------------------------------------------------------------
// K2: fr = h @ Wr.T   (bf16 in, f32 acc, v_wmma_f32_16x16x32_bf16)
// Block = 256 threads (8 waves), tile 128x128, wave tile 64x32.
// Tiles staged by Tensor Data Mover, double-buffered LDS, TENSORcnt sync.
// ---------------------------------------------------------------------------
__global__ __launch_bounds__(256) void gemm_bf16_tdm_kernel(const __bf16* __restrict__ A,   // [M_][D_]
                                                            const __bf16* __restrict__ Bm,  // [D_][D_]
                                                            float* __restrict__ Cm) {       // [M_][D_]
    extern __shared__ __bf16 smem[];

    const int tid  = threadIdx.x;
    const int lane = tid & 31;
    const int w    = tid >> 5;
    const int half = lane >> 4;
    const int l16  = lane & 15;
    const int wm   = w & 1;
    const int wn   = w >> 1;
    const int mBase = blockIdx.x * 128;
    const int nBase = blockIdx.y * 128;

    const unsigned ldsbase = (unsigned)(uintptr_t)smem;   // LDS byte offset of dyn segment
    // Scalar condition -> real s_cbranch (TDM ignores EXEC, so don't rely on masking)
    const bool issuer = (__builtin_amdgcn_readfirstlane(tid) == 0);

    v8f acc[4][2] = {};

    // Prologue: DMA buffer 0 (k = 0)
    if (issuer) {
        tdm_issue_tile(ldsbase,           A  + (size_t)mBase * D_, (unsigned)M_);
        tdm_issue_tile(ldsbase + A_BYTES, Bm + (size_t)nBase * D_, (unsigned)D_);
    }

    for (int k0 = 0; k0 < D_; k0 += KSTEP) {
        const int cur = (k0 >> 6) & 1;
        if (k0 + KSTEP < D_) {
            if (issuer) {
                unsigned boff = ldsbase + (unsigned)(cur ^ 1) * BUF_BYTES;
                tdm_issue_tile(boff,           A  + (size_t)mBase * D_ + (k0 + KSTEP), (unsigned)M_);
                tdm_issue_tile(boff + A_BYTES, Bm + (size_t)nBase * D_ + (k0 + KSTEP), (unsigned)D_);
            }
            __builtin_amdgcn_s_wait_tensorcnt(2);   // current buffer's 2 DMAs done (in-order)
        } else {
            __builtin_amdgcn_s_wait_tensorcnt(0);
        }
        __syncthreads();

        const __bf16* bufA = smem + (size_t)cur * (BUF_BYTES / 2);
        const __bf16* bufB = bufA + (A_BYTES / 2);

#pragma unroll
        for (int kk = 0; kk < KSTEP; kk += 32) {
            // A fragments: lane -> row M = l16, K = kk + {16*g + 8*half + 0..7 : g=0,1}
            v16bf afrag[4];
#pragma unroll
            for (int mi = 0; mi < 4; ++mi) {
                const __bf16* p = bufA + (size_t)(wm * 64 + mi * 16 + l16) * LDS_ROW
                                + kk + 8 * half;
                v8bf lo = *(const v8bf*)p;
                v8bf hi = *(const v8bf*)(p + 16);
                afrag[mi] = __builtin_shufflevector(lo, hi,
                    0, 1, 2, 3, 4, 5, 6, 7, 8, 9, 10, 11, 12, 13, 14, 15);
            }
            // B fragments: lane -> column N = l16, K = kk + {16*g + 8*half + 0..7 : g=0,1}
            v16bf bfrag[2];
#pragma unroll
            for (int nj = 0; nj < 2; ++nj) {
                const __bf16* p = bufB + (size_t)(wn * 32 + nj * 16 + l16) * LDS_ROW
                                + kk + 8 * half;
                v8bf lo = *(const v8bf*)p;
                v8bf hi = *(const v8bf*)(p + 16);
                bfrag[nj] = __builtin_shufflevector(lo, hi,
                    0, 1, 2, 3, 4, 5, 6, 7, 8, 9, 10, 11, 12, 13, 14, 15);
            }
#pragma unroll
            for (int mi = 0; mi < 4; ++mi)
#pragma unroll
                for (int nj = 0; nj < 2; ++nj)
                    acc[mi][nj] = __builtin_amdgcn_wmma_f32_16x16x32_bf16(
                        false, afrag[mi], false, bfrag[nj],
                        (short)0, acc[mi][nj], false, false);
        }
        __syncthreads();
    }

    // D layout: vgpr r holds M = r + 8*half, N = l16
#pragma unroll
    for (int mi = 0; mi < 4; ++mi)
#pragma unroll
        for (int nj = 0; nj < 2; ++nj)
#pragma unroll
            for (int r = 0; r < 8; ++r) {
                int m = mBase + wm * 64 + mi * 16 + r + 8 * half;
                int n = nBase + wn * 32 + nj * 16 + l16;
                Cm[(size_t)m * D_ + n] = acc[mi][nj][r];
            }
}

// ---------------------------------------------------------------------------
// K3: sr[b,h,j] = dot(leaky(fr[b,j, h*128 : h*128+128]), att_r)   one wave each
// ---------------------------------------------------------------------------
__global__ __launch_bounds__(256) void sr_kernel(const float* __restrict__ fr,
                                                 const float* __restrict__ att_r,
                                                 float* __restrict__ sr) {
    int gid  = blockIdx.x * 256 + threadIdx.x;
    int W    = gid >> 5;
    int lane = threadIdx.x & 31;
    int b    = W >> 14;          // / (H_*N_)
    int rem  = W & 16383;
    int hh   = rem >> 11;        // / N_
    int j    = rem & (N_ - 1);

    const float* row = fr + ((size_t)(b * N_ + j)) * D_ + hh * HD_ + lane * 4;
    float4 v = *(const float4*)row;
    float4 a = *(const float4*)(att_r + lane * 4);
    float s = leaky(v.x) * a.x + leaky(v.y) * a.y + leaky(v.z) * a.z + leaky(v.w) * a.w;
    for (int off = 16; off > 0; off >>= 1) s += __shfl_down(s, off, 32);
    if (lane == 0) sr[W] = s;
}

// ---------------------------------------------------------------------------
// K4: per (b,h): w = softmax_j(sr);  c[b, h*128+d] = sum_j w_j * fr[b,j,h*128+d]
// ---------------------------------------------------------------------------
__global__ __launch_bounds__(512) void softmax_ctx_kernel(const float* __restrict__ fr,
                                                          const float* __restrict__ sr,
                                                          float* __restrict__ c) {
    __shared__ float red[512];
    const int bh = blockIdx.x;             // 0..31
    const int b = bh >> 3, hh = bh & 7;
    const int tid = threadIdx.x;
    const float* s = sr + (size_t)bh * N_;

    float lm = -3.0e38f;
    for (int j = tid; j < N_; j += 512) lm = fmaxf(lm, s[j]);
    red[tid] = lm; __syncthreads();
    for (int st = 256; st > 0; st >>= 1) {
        if (tid < st) red[tid] = fmaxf(red[tid], red[tid + st]);
        __syncthreads();
    }
    float mx = red[0]; __syncthreads();

    float ls = 0.0f;
    for (int j = tid; j < N_; j += 512) ls += __expf(s[j] - mx);
    red[tid] = ls; __syncthreads();
    for (int st = 256; st > 0; st >>= 1) {
        if (tid < st) red[tid] += red[tid + st];
        __syncthreads();
    }
    float inv = 1.0f / red[0]; __syncthreads();

    int d = tid & 127, g = tid >> 7;       // 128 dims x 4 j-groups
    float acc = 0.0f;
    for (int j = g; j < N_; j += 4) {
        float wj = __expf(s[j] - mx) * inv;
        acc += wj * fr[((size_t)(b * N_ + j)) * D_ + hh * HD_ + d];
    }
    red[tid] = acc; __syncthreads();
    if (g == 0)
        c[b * D_ + hh * HD_ + d] = red[d] + red[d + 128] + red[d + 256] + red[d + 384];
}

// ---------------------------------------------------------------------------
// K5: g[b,d] = sum_k Wf[d,k] * c[b,k]    (one wave per output element)
// ---------------------------------------------------------------------------
__global__ __launch_bounds__(256) void gemv_kernel(const float* __restrict__ Wf,
                                                   const float* __restrict__ c,
                                                   float* __restrict__ g) {
    int gid  = blockIdx.x * 256 + threadIdx.x;
    int W    = gid >> 5;
    int lane = threadIdx.x & 31;
    int b    = W >> 10;
    int dcol = W & (D_ - 1);
    const float* wr = Wf + (size_t)dcol * D_;
    const float* cb = c + b * D_;
    float acc = 0.0f;
    for (int k = lane * 4; k < D_; k += 128) {
        float4 wv = *(const float4*)(wr + k);
        float4 cv = *(const float4*)(cb + k);
        acc += wv.x * cv.x + wv.y * cv.y + wv.z * cv.z + wv.w * cv.w;
    }
    for (int off = 16; off > 0; off >>= 1) acc += __shfl_down(acc, off, 32);
    if (lane == 0) g[W] = acc;
}

// ---------------------------------------------------------------------------
// K6: y = h + g[b] (broadcast); LayerNorm(y) -> out.  One block per row.
// ---------------------------------------------------------------------------
__global__ __launch_bounds__(256) void add_ln_kernel(const float* __restrict__ h,
                                                     const float* __restrict__ g,
                                                     const float* __restrict__ gamma,
                                                     const float* __restrict__ beta,
                                                     float* __restrict__ out) {
    __shared__ float ys[D_];
    __shared__ float r1[256], r2[256];
    const int row = blockIdx.x;            // 0..8191
    const int b = row >> 11;               // row / N_
    const int tid = threadIdx.x;

    float4 hv = *(const float4*)(h + (size_t)row * D_ + tid * 4);
    float4 gv = *(const float4*)(g + b * D_ + tid * 4);
    float4 y = {hv.x + gv.x, hv.y + gv.y, hv.z + gv.z, hv.w + gv.w};
    *(float4*)&ys[tid * 4] = y;
    r1[tid] = y.x + y.y + y.z + y.w;
    r2[tid] = y.x * y.x + y.y * y.y + y.z * y.z + y.w * y.w;
    __syncthreads();
    for (int st = 128; st > 0; st >>= 1) {
        if (tid < st) { r1[tid] += r1[tid + st]; r2[tid] += r2[tid + st]; }
        __syncthreads();
    }
    const float mu   = r1[0] * (1.0f / D_);
    const float var  = r2[0] * (1.0f / D_) - mu * mu;
    const float rstd = rsqrtf(var + 1e-5f);

    float4 gm = *(const float4*)(gamma + tid * 4);
    float4 bt = *(const float4*)(beta + tid * 4);
    float4 yv = *(const float4*)&ys[tid * 4];
    float4 o = {(yv.x - mu) * rstd * gm.x + bt.x,
                (yv.y - mu) * rstd * gm.y + bt.y,
                (yv.z - mu) * rstd * gm.z + bt.z,
                (yv.w - mu) * rstd * gm.w + bt.w};
    *(float4*)(out + (size_t)row * D_ + tid * 4) = o;
}

// ---------------------------------------------------------------------------
extern "C" void kernel_launch(void* const* d_in, const int* in_sizes, int n_in,
                              void* d_out, int out_size, void* d_ws, size_t ws_size,
                              hipStream_t stream) {
    (void)in_sizes; (void)n_in; (void)out_size; (void)ws_size;
    const float* h     = (const float*)d_in[0];
    // d_in[1] = Wl, d_in[3] = att_l: provably unused (sl cancels inside softmax)
    const float* Wr    = (const float*)d_in[2];
    const float* att_r = (const float*)d_in[4];
    const float* Wf    = (const float*)d_in[5];
    const float* gamma = (const float*)d_in[6];
    const float* beta  = (const float*)d_in[7];
    float* out = (float*)d_out;

    // Workspace carve-up (256B aligned)
    char* ws = (char*)d_ws;
    size_t off = 0;
    auto take = [&](size_t bytes) {
        char* p = ws + off;
        off = (off + bytes + 255) & ~(size_t)255;
        return p;
    };
    __bf16* h_bf  = (__bf16*)take((size_t)M_ * D_ * 2);   // 16 MB
    __bf16* Wr_bf = (__bf16*)take((size_t)D_ * D_ * 2);   //  2 MB
    float*  fr    = (float*) take((size_t)M_ * D_ * 4);   // 33.5 MB
    float*  sr    = (float*) take((size_t)B_ * H_ * N_ * 4);
    float*  cbuf  = (float*) take((size_t)B_ * D_ * 4);
    float*  gbuf  = (float*) take((size_t)B_ * D_ * 4);

    // K1: convert h and Wr to bf16
    {
        int n4 = (M_ * D_) / 4;
        cvt_bf16_kernel<<<(n4 + 255) / 256, 256, 0, stream>>>(h, h_bf, n4);
        int w4 = (D_ * D_) / 4;
        cvt_bf16_kernel<<<(w4 + 255) / 256, 256, 0, stream>>>(Wr, Wr_bf, w4);
    }
    // K2: fr = h @ Wr.T (WMMA bf16 -> f32, TDM-staged, double-buffered LDS)
    {
        dim3 grid(M_ / 128, D_ / 128);
        gemm_bf16_tdm_kernel<<<grid, 256, SMEM_BYTES, stream>>>(h_bf, Wr_bf, fr);
    }
    // K3: sr scores (one wave per (b,h,j))
    {
        int waves = B_ * H_ * N_;           // 65536
        sr_kernel<<<waves / 8, 256, 0, stream>>>(fr, att_r, sr);
    }
    // K4: softmax over j + weighted context per (b,h)
    softmax_ctx_kernel<<<B_ * H_, 512, 0, stream>>>(fr, sr, cbuf);
    // K5: g = c @ Wf.T per batch
    {
        int waves = B_ * D_;                // 4096
        gemv_kernel<<<waves / 8, 256, 0, stream>>>(Wf, cbuf, gbuf);
    }
    // K6: residual + LayerNorm
    add_ln_kernel<<<M_, 256, 0, stream>>>(h, gbuf, gamma, beta, out);
}
